// SimpleGNN_41506563948610
// MI455X (gfx1250) — compile-verified, matrix operand order checked by reading, not run on hardware
//
#include <hip/hip_runtime.h>

typedef __attribute__((ext_vector_type(16))) _Float16 v16h;
typedef __attribute__((ext_vector_type(8)))  _Float16 v8h;
typedef __attribute__((ext_vector_type(8)))  float    v8f;

#define N_NODES 100000
#define N_EDGES 1600000
#define HID 64

// ---------------------------------------------------------------------------
// WMMA helper: D = A(16x32 f16) x B(32x16 f16) + C(16x16 f32)
// ---------------------------------------------------------------------------
__device__ __forceinline__ v8f wmma_f32_f16(v16h a, v16h b, v8f c) {
    return __builtin_amdgcn_wmma_f32_16x16x32_f16(
        /*neg_a=*/false, a, /*neg_b=*/false, b,
        /*c_mod=*/(short)0, c, /*reuse_a=*/false, /*reuse_b=*/false);
}

// Pre-swizzled B fragments live in workspace: frag f = 512 halves, lane L owns
// halves [f*512 + L*16 .. +15] so a fragment load is two contiguous 16B reads.
__device__ __forceinline__ v16h load_frag(const _Float16* __restrict__ frags,
                                          int f, int lane) {
    return *(const v16h*)(frags + (size_t)f * 512 + (size_t)lane * 16);
}

// Fragment table (index -> weight tile):
//  0..3  : W0   (16x64, K padded to 32), ntile = f
//  4..11 : Mtop (M0 rows 0..63),  f = 4  + kt*4 + nt
// 12..19 : Mbot (M0 rows 64..127),f = 12 + kt*4 + nt
// 20..27 : W1   (64x64),          f = 20 + kt*4 + nt
// 28..29 : W2   (64x1, N padded), f = 28 + kt
__global__ __launch_bounds__(32) void prep_frags(
    const float* __restrict__ W0, const float* __restrict__ M0,
    const float* __restrict__ W1, const float* __restrict__ W2,
    _Float16* __restrict__ frags)
{
    int f = blockIdx.x;
    int lane = threadIdx.x;
    const float* W; int krow0, col0, krows, ncols, ld;
    if (f < 4)       { W = W0;            krow0 = 0;            col0 = f * 16;        krows = 16; ncols = 64; ld = 64; }
    else if (f < 12) { int i = f - 4;  W = M0;            krow0 = (i >> 2) * 32; col0 = (i & 3) * 16; krows = 64; ncols = 64; ld = 64; }
    else if (f < 20) { int i = f - 12; W = M0 + 64 * 64;  krow0 = (i >> 2) * 32; col0 = (i & 3) * 16; krows = 64; ncols = 64; ld = 64; }
    else if (f < 28) { int i = f - 20; W = W1;            krow0 = (i >> 2) * 32; col0 = (i & 3) * 16; krows = 64; ncols = 64; ld = 64; }
    else             { int i = f - 28; W = W2;            krow0 = i * 32;        col0 = 0;            krows = 64; ncols = 1;  ld = 1;  }

    // B-matrix (32x16 K-major tile) lane layout: lane holds column n = L&15;
    // halves 0..7 -> K = kh..kh+7, halves 8..15 -> K = 16+kh..16+kh+7, kh=(L>>4)*8
    int n  = lane & 15;
    int kh = (lane >> 4) * 8;
    _Float16* dst = frags + (size_t)f * 512 + (size_t)lane * 16;
    #pragma unroll
    for (int j = 0; j < 16; ++j) {
        int kk = (j < 8) ? (krow0 + kh + j) : (krow0 + 16 + kh + (j - 8));
        int c  = col0 + n;
        float v = (kk < krows && c < ncols) ? W[(size_t)kk * ld + c] : 0.0f;
        dst[j] = (_Float16)v;
    }
}

// ---------------------------------------------------------------------------
// Compute one 16-node tile of h = relu(x@W0 + b0) into C-layout accumulators.
// hv[nt][i] holds element (m = i + 8*(lane>>4), n = nt*16 + (lane&15)).
// ---------------------------------------------------------------------------
__device__ __forceinline__ void compute_h_tile(
    const float* __restrict__ x, const float* __restrict__ b0,
    const _Float16* __restrict__ frags, int base, int lane, v8f hv[4])
{
    int r  = lane & 15;
    int kh = (lane >> 4) * 8;
    v16h ax = {};                       // K=16..31 stays zero (K padded)
    int node = base + r;
    if (node < N_NODES) {
        const float* xr = x + (size_t)node * 16 + kh;
        #pragma unroll
        for (int j = 0; j < 8; ++j) ax[j] = (_Float16)xr[j];
    }
    #pragma unroll
    for (int nt = 0; nt < 4; ++nt) {
        v8f acc = {};
        acc = wmma_f32_f16(ax, load_frag(frags, nt, lane), acc);
        float bias = b0[nt * 16 + (lane & 15)];
        #pragma unroll
        for (int i = 0; i < 8; ++i) {
            float v = acc[i] + bias;
            hv[nt][i] = v > 0.0f ? v : 0.0f;
        }
    }
}

// Load two A fragments (K=0..31, K=32..63) from a row-major [16][64] f16 LDS tile.
__device__ __forceinline__ void load_A_from_lds(
    const _Float16* ldsTile, int lane, v16h a[2])
{
    int r  = lane & 15;
    int kh = (lane >> 4) * 8;
    #pragma unroll
    for (int kt = 0; kt < 2; ++kt) {
        const _Float16* p = ldsTile + r * 64 + kt * 32 + kh;   // 16B aligned
        v8h lo = *(const v8h*)p;
        v8h hi = *(const v8h*)(p + 16);
        #pragma unroll
        for (int j = 0; j < 8; ++j) { a[kt][j] = lo[j]; a[kt][j + 8] = hi[j]; }
    }
}

// ---------------------------------------------------------------------------
// Kernel 1: per 16-node tile, h = relu(x@W0+b0); P = h@Mtop; Q = h@Mbot + bM0
// ---------------------------------------------------------------------------
__global__ __launch_bounds__(256) void node_pq(
    const float* __restrict__ x, const float* __restrict__ b0,
    const float* __restrict__ bM0, const _Float16* __restrict__ frags,
    float* __restrict__ P, float* __restrict__ Q)
{
    __shared__ _Float16 ldsH[8][16 * 64];
    int wave = threadIdx.x >> 5, lane = threadIdx.x & 31;
    int base = (blockIdx.x * 8 + wave) * 16;

    v8f hv[4];
    compute_h_tile(x, b0, frags, base, lane, hv);

    // stage h tile (f16, row-major [16][64]) in LDS for the K=64 GEMMs
    #pragma unroll
    for (int nt = 0; nt < 4; ++nt) {
        int n = nt * 16 + (lane & 15);
        #pragma unroll
        for (int i = 0; i < 8; ++i) {
            int m = i + ((lane >> 4) << 3);
            ldsH[wave][m * 64 + n] = (_Float16)hv[nt][i];
        }
    }
    __syncthreads();

    v16h a[2];
    load_A_from_lds(ldsH[wave], lane, a);

    #pragma unroll
    for (int nt = 0; nt < 4; ++nt) {
        v8f accP = {}, accQ = {};
        accP = wmma_f32_f16(a[0], load_frag(frags,  4 + nt, lane), accP);
        accP = wmma_f32_f16(a[1], load_frag(frags,  8 + nt, lane), accP);
        accQ = wmma_f32_f16(a[0], load_frag(frags, 12 + nt, lane), accQ);
        accQ = wmma_f32_f16(a[1], load_frag(frags, 16 + nt, lane), accQ);
        int n = nt * 16 + (lane & 15);
        float bq = bM0[n];
        #pragma unroll
        for (int i = 0; i < 8; ++i) {
            int m = i + ((lane >> 4) << 3);
            int g = base + m;
            if (g < N_NODES) {
                P[(size_t)g * 64 + n] = accP[i];
                Q[(size_t)g * 64 + n] = accQ[i] + bq;
            }
        }
    }
}

// ---------------------------------------------------------------------------
// Kernel 2: edge scatter. 8 threads/edge, 8 columns each.
// agg[dst] += P[src]; deg[dst] += 1. All data L2-resident (25.6MB buffers).
// ---------------------------------------------------------------------------
__global__ __launch_bounds__(256) void edge_scatter(
    const int* __restrict__ ei, const float* __restrict__ P,
    float* __restrict__ agg, float* __restrict__ deg)
{
    long long t = (long long)blockIdx.x * 256 + threadIdx.x;
    int e = (int)(t >> 3);
    int part = (int)(t & 7);
    if (e >= N_EDGES) return;
    int src = ei[e];
    int dst = ei[N_EDGES + e];

    const float4* ps = (const float4*)(P + (size_t)src * 64 + part * 8);
    float4 a0 = ps[0];
    float4 a1 = ps[1];
    float* ad = agg + (size_t)dst * 64 + part * 8;
    __hip_atomic_fetch_add(ad + 0, a0.x, __ATOMIC_RELAXED, __HIP_MEMORY_SCOPE_AGENT);
    __hip_atomic_fetch_add(ad + 1, a0.y, __ATOMIC_RELAXED, __HIP_MEMORY_SCOPE_AGENT);
    __hip_atomic_fetch_add(ad + 2, a0.z, __ATOMIC_RELAXED, __HIP_MEMORY_SCOPE_AGENT);
    __hip_atomic_fetch_add(ad + 3, a0.w, __ATOMIC_RELAXED, __HIP_MEMORY_SCOPE_AGENT);
    __hip_atomic_fetch_add(ad + 4, a1.x, __ATOMIC_RELAXED, __HIP_MEMORY_SCOPE_AGENT);
    __hip_atomic_fetch_add(ad + 5, a1.y, __ATOMIC_RELAXED, __HIP_MEMORY_SCOPE_AGENT);
    __hip_atomic_fetch_add(ad + 6, a1.z, __ATOMIC_RELAXED, __HIP_MEMORY_SCOPE_AGENT);
    __hip_atomic_fetch_add(ad + 7, a1.w, __ATOMIC_RELAXED, __HIP_MEMORY_SCOPE_AGENT);
    if (part == 0)
        __hip_atomic_fetch_add(deg + dst, 1.0f, __ATOMIC_RELAXED, __HIP_MEMORY_SCOPE_AGENT);
}

// ---------------------------------------------------------------------------
// Kernel 3: recompute h, add normalized message, GEMM1(+relu), GEMM2 -> out
// h2 = h + (scatterP + deg*Q) / max(deg,1);  out = relu(h2@W1+b1) @ W2 + b2
// ---------------------------------------------------------------------------
__global__ __launch_bounds__(256) void node_final(
    const float* __restrict__ x,   const float* __restrict__ b0,
    const float* __restrict__ agg, const float* __restrict__ Q,
    const float* __restrict__ deg, const float* __restrict__ b1,
    const float* __restrict__ b2,  const _Float16* __restrict__ frags,
    float* __restrict__ out)
{
    __shared__ _Float16 ldsA[8][16 * 64];
    __shared__ _Float16 ldsG[8][16 * 64];
    int wave = threadIdx.x >> 5, lane = threadIdx.x & 31;
    int base = (blockIdx.x * 8 + wave) * 16;

    v8f hv[4];
    compute_h_tile(x, b0, frags, base, lane, hv);

    // compose residual + normalized aggregate, stage f16 tile in LDS
    #pragma unroll
    for (int nt = 0; nt < 4; ++nt) {
        int n = nt * 16 + (lane & 15);
        #pragma unroll
        for (int i = 0; i < 8; ++i) {
            int m = i + ((lane >> 4) << 3);
            int g = base + m;
            float v = 0.0f;
            if (g < N_NODES) {
                float d  = deg[g];
                float dn = d < 1.0f ? 1.0f : d;
                v = hv[nt][i] +
                    (agg[(size_t)g * 64 + n] + d * Q[(size_t)g * 64 + n]) / dn;
            }
            ldsA[wave][m * 64 + n] = (_Float16)v;
        }
    }
    __syncthreads();

    v16h a[2];
    load_A_from_lds(ldsA[wave], lane, a);

    // GEMM1: relu(h2 @ W1 + b1) -> ldsG
    #pragma unroll
    for (int nt = 0; nt < 4; ++nt) {
        v8f acc = {};
        acc = wmma_f32_f16(a[0], load_frag(frags, 20 + nt, lane), acc);
        acc = wmma_f32_f16(a[1], load_frag(frags, 24 + nt, lane), acc);
        int n = nt * 16 + (lane & 15);
        float bb = b1[n];
        #pragma unroll
        for (int i = 0; i < 8; ++i) {
            int m = i + ((lane >> 4) << 3);
            float v = acc[i] + bb;
            ldsG[wave][m * 64 + n] = (_Float16)(v > 0.0f ? v : 0.0f);
        }
    }
    __syncthreads();

    // GEMM2: g @ W2 (+ b2); only column 0 of the 16x16 D tile is valid
    v16h ag[2];
    load_A_from_lds(ldsG[wave], lane, ag);
    v8f acc = {};
    acc = wmma_f32_f16(ag[0], load_frag(frags, 28, lane), acc);
    acc = wmma_f32_f16(ag[1], load_frag(frags, 29, lane), acc);
    if ((lane & 15) == 0) {
        float bo = b2[0];
        #pragma unroll
        for (int i = 0; i < 8; ++i) {
            int m = i + ((lane >> 4) << 3);
            int g = base + m;
            if (g < N_NODES) out[g] = acc[i] + bo;
        }
    }
}

// ---------------------------------------------------------------------------
extern "C" void kernel_launch(void* const* d_in, const int* in_sizes, int n_in,
                              void* d_out, int out_size, void* d_ws, size_t ws_size,
                              hipStream_t stream)
{
    (void)in_sizes; (void)n_in; (void)out_size; (void)ws_size;
    const float* x   = (const float*)d_in[0];
    const int*   ei  = (const int*)  d_in[1];
    const float* W0  = (const float*)d_in[2];
    const float* b0  = (const float*)d_in[3];
    const float* W1  = (const float*)d_in[4];
    const float* b1  = (const float*)d_in[5];
    const float* W2  = (const float*)d_in[6];
    const float* b2  = (const float*)d_in[7];
    const float* M0  = (const float*)d_in[8];
    const float* bM0 = (const float*)d_in[9];

    char* wsb = (char*)d_ws;
    _Float16* frags = (_Float16*)wsb;                         // 30*512*2 B < 32 KB
    float* P   = (float*)(wsb + 32768);                       // [N,64]
    float* Q   = P   + (size_t)N_NODES * HID;                 // [N,64]
    float* agg = Q   + (size_t)N_NODES * HID;                 // [N,64]
    float* deg = agg + (size_t)N_NODES * HID;                 // [N]

    hipMemsetAsync(agg, 0, (size_t)N_NODES * HID * sizeof(float), stream);
    hipMemsetAsync(deg, 0, (size_t)N_NODES * sizeof(float), stream);

    prep_frags<<<30, 32, 0, stream>>>(W0, M0, W1, W2, frags);

    int ntiles  = (N_NODES + 15) / 16;        // 6250 wave tiles
    int nblocks = (ntiles + 7) / 8;           // 8 waves / block
    node_pq<<<nblocks, 256, 0, stream>>>(x, b0, bM0, frags, P, Q);

    long long et = (long long)N_EDGES * 8;
    edge_scatter<<<(int)((et + 255) / 256), 256, 0, stream>>>(ei, P, agg, deg);

    node_final<<<nblocks, 256, 0, stream>>>(x, b0, agg, Q, deg, b1, b2, frags,
                                            (float*)d_out);
}
